// EfficientAttention_51316269252743
// MI455X (gfx1250) — compile-verified, matrix operand order checked by reading, not run on hardware
//
#include <hip/hip_runtime.h>
#include <hip/hip_bf16.h>
#include <cstdint>
#include <cstddef>

// ---------------------------------------------------------------------------
// Types / constants
// ---------------------------------------------------------------------------
typedef __attribute__((ext_vector_type(16))) __bf16 v16bf;
typedef __attribute__((ext_vector_type(8)))  float  v8f;
typedef unsigned short ushort_t;

#define B_   4
#define S_   8192
#define H_   512
#define NH_  8
#define HD_  64
#define FD_  32
#define M_   (B_*S_)      // 32768 rows
#define N1_  (3*H_)       // 1536

// ---------------------------------------------------------------------------
// Helpers
// ---------------------------------------------------------------------------
__device__ __forceinline__ ushort_t f2bfbits(float f) {
    unsigned u = __float_as_uint(f);
    unsigned r = (u + 0x7FFFu + ((u >> 16) & 1u)) >> 16;   // RNE
    return (ushort_t)r;
}
__device__ __forceinline__ __bf16 bits2bf(ushort_t b) {
    return __builtin_bit_cast(__bf16, b);
}
__device__ __forceinline__ __bf16 f2bf(float f) { return bits2bf(f2bfbits(f)); }

__device__ __forceinline__ v8f wmma_bf16(v16bf a, v16bf b, v8f c) {
    return __builtin_amdgcn_wmma_f32_16x16x32_bf16(
        /*neg_a=*/false, a, /*neg_b=*/false, b,
        /*c_mod=*/(short)0, c, /*reuse_a=*/false, /*reuse_b=*/false);
}

// 16-bit fragment from LDS/registers when the operand row (fixed M or fixed N)
// is contiguous in K: two contiguous 16B chunks per lane.
// K pattern (ISA 7.12.2): elems 0..7 -> K = 8*hl + e ; elems 8..15 -> K = 16 + 8*hl + e
__device__ __forceinline__ v16bf load_frag16(const ushort_t* rowbase, int hl) {
    struct P { uint4 lo, hi; } p;
    p.lo = *reinterpret_cast<const uint4*>(rowbase + 8 * hl);
    p.hi = *reinterpret_cast<const uint4*>(rowbase + 16 + 8 * hl);
    return __builtin_bit_cast(v16bf, p);
}

// Same fragment shape, sourced from contiguous fp32 (converted on the fly).
__device__ __forceinline__ v16bf load_frag_f32(const float* rowbase, int hl) {
    v16bf f;
    #pragma unroll
    for (int e = 0; e < 8; ++e) {
        f[e]     = f2bf(rowbase[8 * hl + e]);
        f[e + 8] = f2bf(rowbase[16 + 8 * hl + e]);
    }
    return f;
}

// ---------------------------------------------------------------------------
// Conversion / init kernels
// ---------------------------------------------------------------------------
__global__ void cvt4_f32_bf16(const float* __restrict__ in,
                              ushort_t* __restrict__ out, size_t n4) {
    size_t i = (size_t)blockIdx.x * blockDim.x + threadIdx.x;
    if (i < n4) {
        float4 v = reinterpret_cast<const float4*>(in)[i];
        uint2 r;
        r.x = (unsigned)f2bfbits(v.x) | ((unsigned)f2bfbits(v.y) << 16);
        r.y = (unsigned)f2bfbits(v.z) | ((unsigned)f2bfbits(v.w) << 16);
        reinterpret_cast<uint2*>(out)[i] = r;
    }
}

// in: K x N row-major fp32  ->  out: N x K row-major bf16 (transposed weights)
__global__ void cvt_transpose(const float* __restrict__ in,
                              ushort_t* __restrict__ out, int K, int N) {
    int n = blockIdx.x * 32 + (threadIdx.x & 31);
    int k = blockIdx.y * 8 + (threadIdx.x >> 5);
    if (n < N && k < K)
        out[(size_t)n * K + k] = f2bfbits(in[(size_t)k * N + n]);
}

__global__ void zero_f32(float* __restrict__ p, size_t n) {
    size_t i = (size_t)blockIdx.x * blockDim.x + threadIdx.x;
    if (i < n) p[i] = 0.0f;
}

// ---------------------------------------------------------------------------
// Tiled bf16 GEMM (TN): C(f32, MxN) = A(bf16, MxK) * Bt(bf16, NxK)^T [+ bias]
// Block 256 thr = 8 waves; tile 128x128x32; wave tile 32x64 (2x4 WMMA)
// Double-buffered LDS; both A and B fragments are 2 x ds_load_b128.
// ---------------------------------------------------------------------------
#define GBM 128
#define GBN 128
#define GBK 32
#define LDT (GBK + 8)    // 40 shorts (80B row stride, 16B aligned)

__global__ __launch_bounds__(256)
void gemm_bf16_tn(const ushort_t* __restrict__ A, const ushort_t* __restrict__ Bt,
                  float* __restrict__ C, const float* __restrict__ bias,
                  int Mdim, int Ndim, int Kdim) {
    alignas(16) __shared__ ushort_t lA[2][GBM * LDT];
    alignas(16) __shared__ ushort_t lB[2][GBN * LDT];

    const int tid  = threadIdx.x;
    const int lane = tid & 31, wid = tid >> 5;
    const int wm = wid & 3, wn = wid >> 2;          // 4 (M) x 2 (N) waves
    const int tm = blockIdx.y * GBM, tn = blockIdx.x * GBN;
    const int m16 = lane & 15, hl = lane >> 4;

    // staging tasks: 512 uint4 chunks each for A and Bt -> 2 per thread
    const int c0 = tid, c1 = tid + 256;
    const int r0 = c0 >> 2, o0 = (c0 & 3) * 8;
    const int r1 = c1 >> 2, o1 = (c1 & 3) * 8;

    v8f acc[2][4] = {};
    uint4 pa0, pa1, pb0, pb1;

#define GLOAD(K0)                                                              \
    do {                                                                       \
        pa0 = *reinterpret_cast<const uint4*>(A  + (size_t)(tm + r0) * Kdim + (K0) + o0); \
        pa1 = *reinterpret_cast<const uint4*>(A  + (size_t)(tm + r1) * Kdim + (K0) + o1); \
        pb0 = *reinterpret_cast<const uint4*>(Bt + (size_t)(tn + r0) * Kdim + (K0) + o0); \
        pb1 = *reinterpret_cast<const uint4*>(Bt + (size_t)(tn + r1) * Kdim + (K0) + o1); \
    } while (0)
#define SSTORE(BUF)                                                            \
    do {                                                                       \
        *reinterpret_cast<uint4*>(&lA[BUF][r0 * LDT + o0]) = pa0;              \
        *reinterpret_cast<uint4*>(&lA[BUF][r1 * LDT + o1]) = pa1;              \
        *reinterpret_cast<uint4*>(&lB[BUF][r0 * LDT + o0]) = pb0;              \
        *reinterpret_cast<uint4*>(&lB[BUF][r1 * LDT + o1]) = pb1;              \
    } while (0)

    GLOAD(0);
    SSTORE(0);
    __syncthreads();

    int buf = 0;
    for (int k0 = 0; k0 < Kdim; k0 += GBK) {
        const bool more = (k0 + GBK) < Kdim;
        if (more) {
            GLOAD(k0 + GBK);
            if (k0 + 2 * GBK < Kdim) {   // warm L2 for the tile after next
                __builtin_prefetch(A  + (size_t)(tm + r0) * Kdim + k0 + 2 * GBK + o0, 0, 1);
                __builtin_prefetch(Bt + (size_t)(tn + r0) * Kdim + k0 + 2 * GBK + o0, 0, 1);
            }
        }

        v16bf af[2], bfv[4];
        #pragma unroll
        for (int mt = 0; mt < 2; ++mt)
            af[mt] = load_frag16(&lA[buf][(wm * 32 + mt * 16 + m16) * LDT], hl);
        #pragma unroll
        for (int nt = 0; nt < 4; ++nt)
            bfv[nt] = load_frag16(&lB[buf][(wn * 64 + nt * 16 + m16) * LDT], hl);

        #pragma unroll
        for (int mt = 0; mt < 2; ++mt)
            #pragma unroll
            for (int nt = 0; nt < 4; ++nt)
                acc[mt][nt] = wmma_bf16(af[mt], bfv[nt], acc[mt][nt]);

        if (more) SSTORE(buf ^ 1);
        __syncthreads();
        buf ^= 1;
    }
#undef GLOAD
#undef SSTORE

    #pragma unroll
    for (int mt = 0; mt < 2; ++mt)
        #pragma unroll
        for (int nt = 0; nt < 4; ++nt)
            #pragma unroll
            for (int r = 0; r < 8; ++r) {
                int row = tm + wm * 32 + mt * 16 + 8 * hl + r;
                int col = tn + wn * 64 + nt * 16 + m16;
                float v = acc[mt][nt][r];
                if (bias) v += bias[col];
                C[(size_t)row * Ndim + col] = v;
            }
}

// ---------------------------------------------------------------------------
// Feature-map projection: q' = elu(q @ RM)+1, k' = elu(k @ RM)+1
// grid: (S/128, B*NH, 2[q|k]); block 256 = 8 waves, each wave 16 rows
// RM staged transposed (FD x HD) in LDS -> B fragments are 2 x ds_load_b128
// ---------------------------------------------------------------------------
__global__ __launch_bounds__(256)
void proj_kernel(const float* __restrict__ qkv, const float* __restrict__ rm,
                 float* __restrict__ qp, float* __restrict__ kp) {
    alignas(16) __shared__ ushort_t lrmt[FD_ * HD_];   // [n=0..31][k=0..63] bf16
    const int tid = threadIdx.x;
    for (int i = tid; i < HD_ * FD_; i += 256) {
        int k = i / FD_, n = i % FD_;                  // coalesced read of rm
        lrmt[n * HD_ + k] = f2bfbits(rm[i]);
    }
    __syncthreads();

    const int lane = tid & 31, wid = tid >> 5;
    const int m16 = lane & 15, hl = lane >> 4;
    const int b = blockIdx.y / NH_, nh = blockIdx.y % NH_;
    const int which = blockIdx.z;                 // 0 -> q, 1 -> k
    const int s0 = blockIdx.x * 128 + wid * 16;

    const float* Arow =
        qkv + (size_t)(b * S_ + s0 + m16) * N1_ + which * H_ + nh * HD_;

    v16bf a[2], bm[2][2];
    #pragma unroll
    for (int kt = 0; kt < 2; ++kt) {
        a[kt] = load_frag_f32(Arow + kt * 32, hl);
        #pragma unroll
        for (int nt = 0; nt < 2; ++nt)
            bm[kt][nt] = load_frag16(&lrmt[(nt * 16 + m16) * HD_ + kt * 32], hl);
    }

    v8f acc[2] = {};
    #pragma unroll
    for (int kt = 0; kt < 2; ++kt)
        #pragma unroll
        for (int nt = 0; nt < 2; ++nt)
            acc[nt] = wmma_bf16(a[kt], bm[kt][nt], acc[nt]);

    float* outp = which ? kp : qp;
    #pragma unroll
    for (int nt = 0; nt < 2; ++nt)
        #pragma unroll
        for (int r = 0; r < 8; ++r) {
            int m = 8 * hl + r;
            float v = acc[nt][r];
            v = (v > 0.0f) ? (v + 1.0f) : __expf(v);   // elu(v)+1
            outp[(size_t)(b * S_ + s0 + m) * (NH_ * FD_) + nh * FD_ +
                 nt * 16 + m16] = v;
        }
}

// ---------------------------------------------------------------------------
// kv = sum_s k'^T v  (per b,h: 32x64, K=S) ; ksum = sum_s k'
// grid: (S/KVCHUNK, B*NH); block 32 (1 wave); fp32 atomic reduction
// ---------------------------------------------------------------------------
#define KVCHUNK 256
__global__ __launch_bounds__(32)
void kv_kernel(const float* __restrict__ kp, const float* __restrict__ qkv,
               float* __restrict__ kv, float* __restrict__ ksum) {
    const int lane = threadIdx.x;
    const int m16 = lane & 15, hl = lane >> 4;
    const int b = blockIdx.y / NH_, nh = blockIdx.y % NH_;
    const int s0 = blockIdx.x * KVCHUNK;

    v8f acc[2][4] = {};
    float ks = 0.0f;                       // lane owns d = lane (FD=32)

    for (int kk = 0; kk < KVCHUNK; kk += 32) {
        v16bf a[2], bv[4];
        #pragma unroll
        for (int mt = 0; mt < 2; ++mt) {
            int d = mt * 16 + m16;
            #pragma unroll
            for (int e = 0; e < 8; ++e) {
                int klo = 8 * hl + e, khi = 16 + 8 * hl + e;
                a[mt][e] = f2bf(
                    kp[(size_t)(b * S_ + s0 + kk + klo) * (NH_ * FD_) + nh * FD_ + d]);
                a[mt][e + 8] = f2bf(
                    kp[(size_t)(b * S_ + s0 + kk + khi) * (NH_ * FD_) + nh * FD_ + d]);
            }
        }
        #pragma unroll
        for (int nt = 0; nt < 4; ++nt) {
            int f = nt * 16 + m16;
            #pragma unroll
            for (int e = 0; e < 8; ++e) {
                int klo = 8 * hl + e, khi = 16 + 8 * hl + e;
                bv[nt][e] = f2bf(
                    qkv[(size_t)(b * S_ + s0 + kk + klo) * N1_ + 2 * H_ + nh * HD_ + f]);
                bv[nt][e + 8] = f2bf(
                    qkv[(size_t)(b * S_ + s0 + kk + khi) * N1_ + 2 * H_ + nh * HD_ + f]);
            }
        }
        #pragma unroll
        for (int mt = 0; mt < 2; ++mt)
            #pragma unroll
            for (int nt = 0; nt < 4; ++nt)
                acc[mt][nt] = wmma_bf16(a[mt], bv[nt], acc[mt][nt]);

        #pragma unroll
        for (int t = 0; t < 32; ++t)
            ks += kp[(size_t)(b * S_ + s0 + kk + t) * (NH_ * FD_) + nh * FD_ + lane];
    }

    float* kvbase = kv + (size_t)(b * NH_ + nh) * FD_ * HD_;
    #pragma unroll
    for (int mt = 0; mt < 2; ++mt)
        #pragma unroll
        for (int nt = 0; nt < 4; ++nt)
            #pragma unroll
            for (int r = 0; r < 8; ++r) {
                int d = mt * 16 + 8 * hl + r;
                int f = nt * 16 + m16;
                atomicAdd(&kvbase[d * HD_ + f], acc[mt][nt][r]);
            }
    atomicAdd(&ksum[(size_t)(b * NH_ + nh) * FD_ + lane], ks);
}

// ---------------------------------------------------------------------------
// z = (q' @ kv) / (q'.ksum + eps), written as bf16 for the output GEMM
// grid: (S/128, B*NH); block 256 = 8 waves, each wave 16 rows (K=32, 1 step)
// kv staged transposed (HD x FD) in LDS -> B fragments are 2 x ds_load_b128
// ---------------------------------------------------------------------------
__global__ __launch_bounds__(256)
void z_kernel(const float* __restrict__ qp, const float* __restrict__ kv,
              const float* __restrict__ ksum, ushort_t* __restrict__ zb) {
    alignas(16) __shared__ ushort_t lkvt[HD_ * FD_];   // [f=0..63][d=0..31] bf16
    __shared__ float lks[FD_];
    const int tid = threadIdx.x;
    const int b = blockIdx.y / NH_, nh = blockIdx.y % NH_;

    const float* kvbase = kv + (size_t)(b * NH_ + nh) * FD_ * HD_;
    for (int i = tid; i < FD_ * HD_; i += 256) {
        int d = i / HD_, f = i % HD_;                  // coalesced read of kv
        lkvt[f * FD_ + d] = f2bfbits(kvbase[i]);
    }
    if (tid < FD_) lks[tid] = ksum[(size_t)(b * NH_ + nh) * FD_ + tid];
    __syncthreads();

    const int lane = tid & 31, wid = tid >> 5;
    const int m16 = lane & 15, hl = lane >> 4;
    const int s0 = blockIdx.x * 128 + wid * 16;

    const float* qrow = qp + (size_t)(b * S_ + s0 + m16) * (NH_ * FD_) + nh * FD_;

    v16bf a = load_frag_f32(qrow, hl);
    v16bf bk[4];
    #pragma unroll
    for (int nt = 0; nt < 4; ++nt)
        bk[nt] = load_frag16(&lkvt[(nt * 16 + m16) * FD_], hl);

    v8f acc[4] = {};
    #pragma unroll
    for (int nt = 0; nt < 4; ++nt) acc[nt] = wmma_bf16(a, bk[nt], acc[nt]);

    // normalizer for row (lane & 15): dot(q'_row, ksum)
    float nv = 0.0f;
    #pragma unroll
    for (int d = 0; d < FD_; ++d) nv += qrow[d] * lks[d];

    #pragma unroll
    for (int r = 0; r < 8; ++r) {
        int m = 8 * hl + r;
        float nm = __shfl(nv, m, 32);
        float sc = 1.0f / (nm + 1e-6f);
        #pragma unroll
        for (int nt = 0; nt < 4; ++nt) {
            zb[(size_t)(b * S_ + s0 + m) * H_ + nh * HD_ + nt * 16 + m16] =
                f2bfbits(acc[nt][r] * sc);
        }
    }
}

// ---------------------------------------------------------------------------
// Host launcher
// ---------------------------------------------------------------------------
extern "C" void kernel_launch(void* const* d_in, const int* in_sizes, int n_in,
                              void* d_out, int out_size, void* d_ws, size_t ws_size,
                              hipStream_t stream) {
    const float* x    = (const float*)d_in[0];   // (B,S,H)
    const float* Wqkv = (const float*)d_in[1];   // (H,3H)
    const float* RM   = (const float*)d_in[2];   // (HD,FD)
    const float* Wout = (const float*)d_in[3];   // (H,H)
    const float* bout = (const float*)d_in[4];   // (H,)
    float* out = (float*)d_out;

    char* ws = (char*)d_ws;
    size_t off = 0;
    auto carve = [&](size_t bytes) -> char* {
        char* p = ws + off;
        off += (bytes + 255) & ~(size_t)255;
        return p;
    };
    ushort_t* xb     = (ushort_t*)carve((size_t)M_ * H_ * 2);
    ushort_t* wqkvT  = (ushort_t*)carve((size_t)H_ * N1_ * 2);  // (3H x H) bf16
    ushort_t* woutT  = (ushort_t*)carve((size_t)H_ * H_ * 2);   // (H x H) bf16, transposed
    float*    qkv    = (float*)carve((size_t)M_ * N1_ * 4);
    float*    qp     = (float*)carve((size_t)M_ * NH_ * FD_ * 4);
    float*    kp     = (float*)carve((size_t)M_ * NH_ * FD_ * 4);
    float*    kvb    = (float*)carve((size_t)B_ * NH_ * FD_ * HD_ * 4);
    float*    ksum   = (float*)carve((size_t)B_ * NH_ * FD_ * 4);
    ushort_t* zb     = (ushort_t*)carve((size_t)M_ * H_ * 2);

    // 1) conversions (weights transposed to N x K for A-shaped B fragments)
    {
        size_t n4 = (size_t)M_ * H_ / 4;
        cvt4_f32_bf16<<<(unsigned)((n4 + 255) / 256), 256, 0, stream>>>(x, xb, n4);
        cvt_transpose<<<dim3(N1_ / 32, H_ / 8), 256, 0, stream>>>(Wqkv, wqkvT, H_, N1_);
        cvt_transpose<<<dim3(H_ / 32, H_ / 8), 256, 0, stream>>>(Wout, woutT, H_, H_);
    }
    // 2) zero kv / ksum accumulators
    {
        size_t n = (size_t)B_ * NH_ * FD_ * HD_;
        zero_f32<<<(unsigned)((n + 255) / 256), 256, 0, stream>>>(kvb, n);
        n = (size_t)B_ * NH_ * FD_;
        zero_f32<<<(unsigned)((n + 255) / 256), 256, 0, stream>>>(ksum, n);
    }
    // 3) qkv = x @ W_qkv        (32768 x 1536 x 512)
    gemm_bf16_tn<<<dim3(N1_ / GBN, M_ / GBM), 256, 0, stream>>>(
        xb, wqkvT, qkv, nullptr, M_, N1_, H_);
    // 4) q' / k' feature maps
    proj_kernel<<<dim3(S_ / 128, B_ * NH_, 2), 256, 0, stream>>>(qkv, RM, qp, kp);
    // 5) kv & ksum reductions over S
    kv_kernel<<<dim3(S_ / KVCHUNK, B_ * NH_), 32, 0, stream>>>(kp, qkv, kvb, ksum);
    // 6) z (normalized), stored bf16
    z_kernel<<<dim3(S_ / 128, B_ * NH_), 256, 0, stream>>>(qp, kvb, ksum, zb);
    // 7) out = z @ W_out + b    (32768 x 512 x 512)
    gemm_bf16_tn<<<dim3(H_ / GBN, M_ / GBM), 256, 0, stream>>>(
        zb, woutT, out, bout, M_, H_, H_);
}
